// SparseMultiHeadAttention_9577777070224
// MI455X (gfx1250) — matrix operand order repaired and checked by reading, not compiled, not run on hardware
//
#include <hip/hip_runtime.h>

// ---------------------------------------------------------------------------
// Sparse multi-head attention, MI455X (gfx1250, wave32, WMMA + TDM).
//   B=2, S=2048, H=1024, NH=16, HD=64.
// The virtual token is mathematically dead in the reference (query row sliced
// off, key/value column fully masked), so it is skipped entirely.
// Pipeline: f32->bf16 cvt; Q/K/V projections via bf16 WMMA GEMM with
// TDM (tensor_load_to_lds) double-buffered tile staging; fused
// flash-attention (online softmax) with v_wmma_f32_16x16x32_bf16; output
// projection back to f32.
// ---------------------------------------------------------------------------

#define BB 2
#define SS 2048
#define HH 1024
#define NHEAD 16
#define HD 64
#define BS (BB * SS)   // 4096 rows

typedef __attribute__((ext_vector_type(16))) __bf16 v16bf;
typedef __attribute__((ext_vector_type(8)))  __bf16 v8bf;
typedef __attribute__((ext_vector_type(8)))  float  v8f;

#ifndef __has_builtin
#define __has_builtin(x) 0
#endif
#if __has_builtin(__builtin_amdgcn_tensor_load_to_lds) && \
    __has_builtin(__builtin_amdgcn_s_wait_tensorcnt)
#define USE_TDM 1
#else
#define USE_TDM 0
#endif

static __device__ __forceinline__ unsigned short f2bf(float f) {
  unsigned u = __float_as_uint(f);
  unsigned r = u + 0x7FFFu + ((u >> 16) & 1u);   // round-to-nearest-even
  return (unsigned short)(r >> 16);
}

static __device__ __forceinline__ v8f wmma_bf16(v16bf a, v16bf b, v8f c) {
  // D = A(16x32 bf16) * B(32x16 bf16) + C(16x16 f32)
  return __builtin_amdgcn_wmma_f32_16x16x32_bf16(
      /*neg_a=*/false, a, /*neg_b=*/false, b,
      /*c_mod=*/(short)0, c, /*reuse_a=*/false, /*reuse_b=*/false);
}

static __device__ __forceinline__ v16bf cat8(v8bf lo, v8bf hi) {
  return __builtin_shufflevector(lo, hi, 0, 1, 2, 3, 4, 5, 6, 7,
                                 8, 9, 10, 11, 12, 13, 14, 15);
}

// TDM writes LDS behind the compiler's back (the LDS address is an integer in
// the descriptor).  Escape the pointer into opaque asm with a memory clobber
// so alias analysis cannot fold the subsequent ds_loads to undef.
static __device__ __forceinline__ void lds_escape(void* p) {
  asm volatile("" : : "v"((unsigned long long)(size_t)p) : "memory");
}

#if USE_TDM
typedef unsigned int u32x4 __attribute__((ext_vector_type(4)));
typedef int          i32x4 __attribute__((ext_vector_type(4)));
typedef int          i32x8 __attribute__((ext_vector_type(8)));

// TDM descriptor per CDNA5 ISA: load a 64-row x 32-col
// bf16 tile (row stride HH elements) from global into LDS at byte lds_byte.
// 2D tensor -> descriptor groups 2/3 zero; no padding; no multicast.
static __device__ __forceinline__ void tdm_load_tile(const void* gptr,
                                                     unsigned lds_byte,
                                                     unsigned rows_total) {
  unsigned long long ga = (unsigned long long)gptr;
  u32x4 g0 = {1u,                                    // count=1, user mode
              lds_byte,                              // LDS byte address
              (unsigned)(ga & 0xFFFFFFFFu),          // global addr lo
              (unsigned)((ga >> 32) & 0x01FFFFFFu) | (2u << 30)};  // hi|type=2
  i32x8 g1;
  g1[0] = 1 << 16;                                   // data_size=1 (2 bytes)
  g1[1] = (int)(((unsigned)HH & 0xFFFFu) << 16);     // tensor_dim0[15:0]
  g1[2] = (int)((((unsigned)HH >> 16) & 0xFFFFu) |   // tensor_dim0[31:16]
                ((rows_total & 0xFFFFu) << 16));     // tensor_dim1[15:0]
  g1[3] = (int)(((rows_total >> 16) & 0xFFFFu) |     // tensor_dim1[31:16]
                (32u << 16));                        // tile_dim0 = 32
  g1[4] = 64;                                        // tile_dim1=64, dim2=0
  g1[5] = HH;                                        // dim0 stride lo32
  g1[6] = 0;                                         // stride hi / dim1 stride
  g1[7] = 0;
  i32x4 z4 = {0, 0, 0, 0};
#if __clang_major__ >= 23
  i32x8 z8 = {0, 0, 0, 0, 0, 0, 0, 0};
  __builtin_amdgcn_tensor_load_to_lds(g0, g1, z4, z4, z8, 0);
#else
  __builtin_amdgcn_tensor_load_to_lds(g0, g1, z4, z4, 0);
#endif
}
#endif  // USE_TDM

// ---------------------------------------------------------------------------
// f32 -> bf16 elementwise convert
// ---------------------------------------------------------------------------
__global__ void cvt_f32_bf16(const float* __restrict__ src,
                             unsigned short* __restrict__ dst, int n) {
  int i = blockIdx.x * blockDim.x + threadIdx.x;
  if (i < n) dst[i] = f2bf(src[i]);
}

// ---------------------------------------------------------------------------
// Per-K-step MAC: A frag from LDS (row = 16w+lane15, K halves per lane half),
// 4 B frags (col = 16nt+lane15, K contiguous), 4 WMMAs.
// ---------------------------------------------------------------------------
static __device__ __forceinline__ void gemm_tile_mac(
    const unsigned short* As, const unsigned short* Bs,
    int w, int lane15, int hl, v8f c[4]) {
  const unsigned short* ap = &As[(16 * w + lane15) * 32 + 16 * hl];
  v16bf af = cat8(*(const v8bf*)ap, *(const v8bf*)(ap + 8));
#pragma unroll
  for (int nt = 0; nt < 4; ++nt) {
    const unsigned short* bp = &Bs[(16 * nt + lane15) * 32 + 16 * hl];
    c[nt] = wmma_bf16(af, *(const v16bf*)bp, c[nt]);
  }
}

// ---------------------------------------------------------------------------
// GEMM: C[M=4096, N=1024] = A[4096,1024](bf16) @ W[N,K](bf16, row-major == Bᵀ
// in WMMA layout) + bias.  64x64 tile per 128-thread block, 4 waves, wave w
// owns rows [16w,16w+16).  Tile staging: TDM double-buffer (wave 0 issues the
// DMA, TENSORcnt=2 keeps one tile in flight behind the WMMAs) with a
// synchronous VGPR-staged fallback if the TDM builtin is unavailable.
// mode: 0 = bf16 head-major [b,h,s,d] (Q/K), 1 = bf16 transposed [b,h,d,s]
// (V), 2 = f32 row-major (final output).
// ---------------------------------------------------------------------------
__global__ __launch_bounds__(128) void gemm_bf16(
    const unsigned short* __restrict__ A,   // [4096][1024] bf16 bits
    const unsigned short* __restrict__ W,   // [1024][1024] bf16 bits
    const float* __restrict__ bias,         // [1024]
    void* __restrict__ out, int mode) {
  // buffer b: A tile at byte b*8192, B tile at byte b*8192+4096 (offset from
  // LDS base 0 -- single shared object).
  __shared__ unsigned short lds[8192];

  const int t = threadIdx.x;
  const int w = t >> 5;          // wave 0..3
  const int lane15 = t & 15;
  const int hl = (t >> 4) & 1;   // lane half (0: lanes 0-15, 1: lanes 16-31)
  const int m0 = blockIdx.y * 64;
  const int n0 = blockIdx.x * 64;

  v8f c[4];
#pragma unroll
  for (int i = 0; i < 4; ++i) c[i] = {};

#if USE_TDM
  if (w == 0) {   // wave-uniform: one DMA per tile, EXEC-independent
    tdm_load_tile(A + (size_t)m0 * HH, 0u, (unsigned)BS);
    tdm_load_tile(W + (size_t)n0 * HH, 4096u, (unsigned)HH);
  }
  for (int it = 0; it < HH / 32; ++it) {
    const int cur = it & 1;
    if (it + 1 < HH / 32) {
      if (w == 0) {
        const unsigned nxtb = (unsigned)(1 - cur) * 8192u;
        tdm_load_tile(A + (size_t)m0 * HH + (it + 1) * 32, nxtb, (unsigned)BS);
        tdm_load_tile(W + (size_t)n0 * HH + (it + 1) * 32, nxtb + 4096u,
                      (unsigned)HH);
        __builtin_amdgcn_s_wait_tensorcnt(2);   // current buffer's pair done
      }
    } else if (w == 0) {
      __builtin_amdgcn_s_wait_tensorcnt(0);     // drain tail
    }
    lds_escape(lds);   // TDM wrote LDS: force the compiler to reload fragments
    __syncthreads();
    gemm_tile_mac(lds + cur * 4096, lds + cur * 4096 + 2048, w, lane15, hl, c);
    __syncthreads();   // compute done before this buffer is refilled
  }
#else
  // fallback: synchronous cooperative staging (thread loads 32B per tile)
  const int lrow = t >> 1;           // 0..63
  const int lc16 = (t & 1) * 16;     // 0 or 16
  for (int k0 = 0; k0 < HH; k0 += 32) {
    __syncthreads();
    {
      const unsigned short* ga = A + (size_t)(m0 + lrow) * HH + k0 + lc16;
      const unsigned short* gb = W + (size_t)(n0 + lrow) * HH + k0 + lc16;
      uint4 a0 = *(const uint4*)ga;
      uint4 a1 = *(const uint4*)(ga + 8);
      uint4 b0 = *(const uint4*)gb;
      uint4 b1 = *(const uint4*)(gb + 8);
      *(uint4*)&lds[lrow * 32 + lc16]            = a0;
      *(uint4*)&lds[lrow * 32 + lc16 + 8]        = a1;
      *(uint4*)&lds[2048 + lrow * 32 + lc16]     = b0;
      *(uint4*)&lds[2048 + lrow * 32 + lc16 + 8] = b1;
    }
    __syncthreads();
    gemm_tile_mac(lds, lds + 2048, w, lane15, hl, c);
  }
#endif

#pragma unroll
  for (int nt = 0; nt < 4; ++nt) {
    const int n = n0 + 16 * nt + lane15;
    const float bv = bias[n];
#pragma unroll
    for (int r = 0; r < 8; ++r) {
      const int m = m0 + 16 * w + r + 8 * hl;   // C layout: reg r -> row r+8*hl
      const float val = c[nt][r] + bv;
      if (mode == 2) {
        ((float*)out)[(size_t)m * HH + n] = val;
      } else {
        const int b = m >> 11, s = m & (SS - 1);
        const int h = n >> 6, d = n & (HD - 1);
        unsigned short bb = f2bf(val);
        if (mode == 0)  // [b,h,s,d]
          ((unsigned short*)out)[(((size_t)(b * NHEAD + h)) * SS + s) * HD + d] = bb;
        else            // [b,h,d,s]  (V transposed so PV B-operand is contiguous)
          ((unsigned short*)out)[(((size_t)(b * NHEAD + h)) * HD + d) * SS + s] = bb;
      }
    }
  }
}

// ---------------------------------------------------------------------------
// Fused sparse attention, flash style.  Block = (64-query tile, head, batch),
// 128 threads / 4 waves; wave w owns query rows [q0+16w, q0+16w+16).
// ---------------------------------------------------------------------------
__global__ __launch_bounds__(128) void attn_kernel(
    const unsigned short* __restrict__ q,     // [b,h,s,d] bf16
    const unsigned short* __restrict__ k,     // [b,h,s,d] bf16
    const unsigned short* __restrict__ v,     // [b,h,d,s] bf16
    const unsigned char* __restrict__ mask,   // [b,h,s,s] bool
    unsigned short* __restrict__ ctx) {       // [b,s,h*d] bf16
  const int t = threadIdx.x;
  const int w = t >> 5;
  const int lane15 = t & 15;
  const int hl = (t >> 4) & 1;
  const int q0 = blockIdx.x * 64;
  const int h  = blockIdx.y;
  const int b  = blockIdx.z;
  const int bh = b * NHEAD + h;

  __shared__ unsigned short P[4][16 * 64];    // per-wave prob tile (bf16)

  // ---- Q fragments (A operand), resident in registers for the whole loop
  const unsigned short* qbase =
      q + ((size_t)bh * SS + (q0 + 16 * w + lane15)) * HD;
  v16bf qf[2];
#pragma unroll
  for (int ks = 0; ks < 2; ++ks) {
    const unsigned short* qp = qbase + ks * 32 + 16 * hl;
    qf[ks] = cat8(*(const v8bf*)qp, *(const v8bf*)(qp + 8));
  }

  const float scale = 0.125f;                 // 1/sqrt(HD)
  float mrow[8], lrow[8];
  v8f acc[4];
#pragma unroll
  for (int i = 0; i < 4; ++i) acc[i] = {};
#pragma unroll
  for (int r = 0; r < 8; ++r) { mrow[r] = -1.0e30f; lrow[r] = 0.0f; }

  const unsigned char* mbase =
      mask + ((size_t)bh * SS + (q0 + 16 * w)) * SS;   // + row*S + col

  for (int kb = 0; kb < SS; kb += 64) {
    if (kb + 64 < SS)
      __builtin_prefetch(mbase + (size_t)(8 * hl) * SS + kb + 64, 0, 1);

    // ---- scores S = Q Kᵀ  (K-dim = HD = 64 -> 2 WMMA steps per N tile)
    v8f s[4];
#pragma unroll
    for (int nt = 0; nt < 4; ++nt) {
      s[nt] = {};
#pragma unroll
      for (int ks = 0; ks < 2; ++ks) {
        const unsigned short* kp =
            k + ((size_t)bh * SS + (kb + 16 * nt + lane15)) * HD +
            ks * 32 + 16 * hl;
        v16bf kf = *(const v16bf*)kp;
        s[nt] = wmma_bf16(qf[ks], kf, s[nt]);
      }
    }

    // ---- mask + scale (masked -> large negative finite; avoids inf-inf NaN)
#pragma unroll
    for (int nt = 0; nt < 4; ++nt) {
#pragma unroll
      for (int r = 0; r < 8; ++r) {
        const int col = kb + 16 * nt + lane15;
        const unsigned char mk = mbase[(size_t)(r + 8 * hl) * SS + col];
        s[nt][r] = mk ? s[nt][r] * scale : -3.0e38f;
      }
    }

    // ---- online softmax: row max (16-lane groups match C-matrix layout)
    float mnew[8], alpha[8], rs[8];
#pragma unroll
    for (int r = 0; r < 8; ++r) {
      float mx = fmaxf(fmaxf(s[0][r], s[1][r]), fmaxf(s[2][r], s[3][r]));
#pragma unroll
      for (int d = 1; d < 16; d <<= 1) mx = fmaxf(mx, __shfl_xor(mx, d, 32));
      mnew[r]  = fmaxf(mrow[r], mx);
      alpha[r] = __expf(mrow[r] - mnew[r]);
    }
#pragma unroll
    for (int r = 0; r < 8; ++r) {
      float sum = 0.0f;
#pragma unroll
      for (int nt = 0; nt < 4; ++nt) {
        float p = __expf(s[nt][r] - mnew[r]);   // masked: exp(-3e38) == 0
        s[nt][r] = p;
        sum += p;
      }
#pragma unroll
      for (int d = 1; d < 16; d <<= 1) sum += __shfl_xor(sum, d, 32);
      rs[r] = sum;
    }
#pragma unroll
    for (int r = 0; r < 8; ++r) {
      lrow[r] = lrow[r] * alpha[r] + rs[r];
      mrow[r] = mnew[r];
    }
#pragma unroll
    for (int nt = 0; nt < 4; ++nt)
#pragma unroll
      for (int r = 0; r < 8; ++r) acc[nt][r] *= alpha[r];

    // ---- P (C layout) -> LDS -> A-layout fragments
#pragma unroll
    for (int nt = 0; nt < 4; ++nt)
#pragma unroll
      for (int r = 0; r < 8; ++r)
        P[w][(r + 8 * hl) * 64 + 16 * nt + lane15] = f2bf(s[nt][r]);
    __syncthreads();

    // ---- acc += P @ V   (K-dim = 64 keys -> 2 WMMA steps per HD tile)
#pragma unroll
    for (int nt = 0; nt < 4; ++nt) {
#pragma unroll
      for (int ks = 0; ks < 2; ++ks) {
        const unsigned short* pp = &P[w][lane15 * 64 + ks * 32 + 16 * hl];
        v16bf pf = cat8(*(const v8bf*)pp, *(const v8bf*)(pp + 8));
        const unsigned short* vp =
            v + ((size_t)bh * HD + 16 * nt + lane15) * SS +
            kb + ks * 32 + 16 * hl;
        v16bf vf = *(const v16bf*)vp;
        acc[nt] = wmma_bf16(pf, vf, acc[nt]);
      }
    }
    __syncthreads();
  }

  // ---- normalize and emit ctx[b, s, h*64 + d] (bf16, input to Wo GEMM)
#pragma unroll
  for (int nt = 0; nt < 4; ++nt) {
#pragma unroll
    for (int r = 0; r < 8; ++r) {
      const int srow = q0 + 16 * w + r + 8 * hl;
      const float o = acc[nt][r] / fmaxf(lrow[r], 1.0e-20f);
      ctx[((size_t)b * SS + srow) * HH + h * HD + 16 * nt + lane15] = f2bf(o);
    }
  }
}

// ---------------------------------------------------------------------------
extern "C" void kernel_launch(void* const* d_in, const int* in_sizes, int n_in,
                              void* d_out, int out_size, void* d_ws,
                              size_t ws_size, hipStream_t stream) {
  (void)in_sizes; (void)n_in; (void)out_size; (void)ws_size;
  const float* x  = (const float*)d_in[0];
  const float* Wq = (const float*)d_in[1];
  const float* bq = (const float*)d_in[2];
  const float* Wk = (const float*)d_in[3];
  const float* bk = (const float*)d_in[4];
  const float* Wv = (const float*)d_in[5];
  const float* bv = (const float*)d_in[6];
  const float* Wo = (const float*)d_in[7];
  const float* bo = (const float*)d_in[8];
  // d_in[9] = virtual_token: mathematically dead (query row sliced off,
  // key/value column fully masked) -> skipped.
  const unsigned char* mask = (const unsigned char*)d_in[10];

  char* ws = (char*)d_ws;
  size_t off = 0;
  auto alloc = [&](size_t bytes) {
    void* p = ws + off;
    off += (bytes + 255) & ~(size_t)255;
    return p;
  };
  unsigned short* xbf  = (unsigned short*)alloc((size_t)BS * HH * 2);
  unsigned short* wqb  = (unsigned short*)alloc((size_t)HH * HH * 2);
  unsigned short* wkb  = (unsigned short*)alloc((size_t)HH * HH * 2);
  unsigned short* wvb  = (unsigned short*)alloc((size_t)HH * HH * 2);
  unsigned short* wob  = (unsigned short*)alloc((size_t)HH * HH * 2);
  unsigned short* qbuf = (unsigned short*)alloc((size_t)BS * HH * 2);
  unsigned short* kbuf = (unsigned short*)alloc((size_t)BS * HH * 2);
  unsigned short* vbuf = (unsigned short*)alloc((size_t)BS * HH * 2);
  unsigned short* cbuf = (unsigned short*)alloc((size_t)BS * HH * 2);

  const int nx = BS * HH, nw = HH * HH;
  cvt_f32_bf16<<<(nx + 255) / 256, 256, 0, stream>>>(x,  xbf, nx);
  cvt_f32_bf16<<<(nw + 255) / 256, 256, 0, stream>>>(Wq, wqb, nw);
  cvt_f32_bf16<<<(nw + 255) / 256, 256, 0, stream>>>(Wk, wkb, nw);
  cvt_f32_bf16<<<(nw + 255) / 256, 256, 0, stream>>>(Wv, wvb, nw);
  cvt_f32_bf16<<<(nw + 255) / 256, 256, 0, stream>>>(Wo, wob, nw);

  dim3 ggrid(HH / 64, BS / 64);   // (16, 64)
  gemm_bf16<<<ggrid, 128, 0, stream>>>(xbf, wqb, bq, qbuf, 0);
  gemm_bf16<<<ggrid, 128, 0, stream>>>(xbf, wkb, bk, kbuf, 0);
  gemm_bf16<<<ggrid, 128, 0, stream>>>(xbf, wvb, bv, vbuf, 1);

  attn_kernel<<<dim3(SS / 64, NHEAD, BB), 128, 0, stream>>>(
      qbuf, kbuf, vbuf, mask, cbuf);

  gemm_bf16<<<ggrid, 128, 0, stream>>>(cbuf, wob, bo, d_out, 2);
}
